// HamiltonianMetric_10909216932439
// MI455X (gfx1250) — matrix operand order repaired and checked by reading, not compiled, not run on hardware
//
#include <hip/hip_runtime.h>
#include <hip/hip_bf16.h>
#include <math.h>

// ---------------------------------------------------------------------------
// HamiltonianMetric fused kernel for gfx1250 (MI455X, wave32, WMMA).
//   h1 = tanh(x @ W1 + b1); h2 = tanh(h1 @ W2 + b2); t = tanh(h2 @ W3 + b3)
//   U = scatter(t) upper-triangular 64x64;  y = U^T @ U
// Strategy: 16-row batch tile per workgroup, all intermediates in LDS,
// f16 WMMA (16x16x32) everywhere, weights pre-transposed to f16 in d_ws,
// native v_tanh_f32 activations, scalarized per-wave work distribution,
// dual accumulators to break WMMA->WMMA dependency chains.
// ---------------------------------------------------------------------------

typedef __attribute__((ext_vector_type(16))) _Float16 v16h;
typedef __attribute__((ext_vector_type(8)))  _Float16 v8h;
typedef __attribute__((ext_vector_type(8)))  float    v8f;

#define BDIM 64
#define HDIM 256
#define TDIM 2080
#define BT   16     // batch rows per workgroup
#define NWAVE 8

// Branch-free tanh: native v_tanh_f32 on gfx1250 if the builtin exists,
// otherwise exp2/rcp formulation (v_exp_f32 + v_rcp_f32, no EXEC divergence).
__device__ __forceinline__ float fast_tanh(float v) {
#if __has_builtin(__builtin_amdgcn_tanhf)
  return __builtin_amdgcn_tanhf(v);
#else
  float xc = fminf(fmaxf(v, -9.0f), 9.0f);
  float e  = __builtin_amdgcn_exp2f(xc * 2.8853900817779268f);  // 2*log2(e)
  return 1.0f - 2.0f * __builtin_amdgcn_rcpf(e + 1.0f);
#endif
}

__device__ __forceinline__ v8f wmma16(v16h a, v16h b, v8f c) {
  // v_wmma_f32_16x16x32_f16: (neg_a, A, neg_b, B, c_mod, C, reuse_a, reuse_b)
  return __builtin_amdgcn_wmma_f32_16x16x32_f16(false, a, false, b,
                                                (short)0, c, false, false);
}

// A-tile 16x32 from row-major f16 buffer (row stride `ld` halfs), K offset k0.
// ISA 7.12.2: lanes 0-15 hold M=lane, K chunks [k0..k0+7] and [k0+16..k0+23];
// lanes 16-31 hold the +8 K offset.  Two contiguous 16B loads per lane.
__device__ __forceinline__ v16h load_a(const _Float16* base, int ld, int k0, int lane) {
  int m  = lane & 15;
  int kb = k0 + ((lane >> 4) << 3);
  const _Float16* p = base + m * ld + kb;
  v8h lo = *(const v8h*)(p);
  v8h hi = *(const v8h*)(p + 16);
  v16h a;
#pragma unroll
  for (int i = 0; i < 8; ++i) { a[i] = lo[i]; a[i + 8] = hi[i]; }
  return a;
}

// B-tile 32x16 from K-major (transposed) f16 storage: row n holds K
// contiguously with stride `ld`.  Lanes 0-15: N=lane, K=[k0..k0+7] and
// [k0+16..k0+23]; lanes 16-31: the +8 K offset.  Two 16B loads per lane.
__device__ __forceinline__ v16h load_b(const _Float16* baseT, int ld, int n0, int k0, int lane) {
  int n  = n0 + (lane & 15);
  int kb = k0 + ((lane >> 4) << 3);
  const _Float16* p = baseT + n * ld + kb;
  v8h lo = *(const v8h*)(p);
  v8h hi = *(const v8h*)(p + 16);
  v16h b;
#pragma unroll
  for (int i = 0; i < 8; ++i) { b[i] = lo[i]; b[i + 8] = hi[i]; }
  return b;
}

// Store a C/D accumulator tile (bias + tanh + cvt f16) into a row-major LDS
// buffer.  C layout: VGPR r of lane l holds (M = r + 8*(l>=16), N = l%16).
__device__ __forceinline__ void store_h(_Float16* hb, int ld, int n0, v8f acc,
                                        const float* __restrict__ bias, int lane) {
  int n  = n0 + (lane & 15);
  float bv = bias[n];
  int mb = (lane >> 4) << 3;
#pragma unroll
  for (int r = 0; r < 8; ++r)
    hb[(mb + r) * ld + n] = (_Float16)fast_tanh(acc[r] + bv);
}

// Weight transpose + f32->f16 convert: Wt[n*K + k] = (f16)W[k*N + n]
__global__ void cvt_transpose_f16(const float* __restrict__ W,
                                  _Float16* __restrict__ Wt, int K, int N) {
  int idx = blockIdx.x * blockDim.x + threadIdx.x;
  if (idx >= K * N) return;
  int n = idx / K;
  int k = idx - n * K;
  Wt[idx] = (_Float16)W[k * N + n];
}

__global__ __launch_bounds__(256)
void hm_fused(const float* __restrict__ x,
              const _Float16* __restrict__ W1t,  // [256][64]
              const float* __restrict__ b1,
              const _Float16* __restrict__ W2t,  // [256][256]
              const float* __restrict__ b2,
              const _Float16* __restrict__ W3t,  // [2080][256]
              const float* __restrict__ b3,
              float* __restrict__ y) {
  __shared__ __align__(16) _Float16 xs[BT * BDIM];        //   2 KB
  __shared__ __align__(16) _Float16 h1[BT * HDIM];        //   8 KB
  __shared__ __align__(16) _Float16 h2[BT * HDIM];        //   8 KB
  __shared__ __align__(16) _Float16 Ut[BT * BDIM * BDIM]; // 128 KB (U^T per row)
  __shared__ unsigned short ijtab[TDIM];                  //   4 KB (j<<8 | i)

  const int tid  = threadIdx.x;
  // Scalarize the wave id: uniform within a wave -> SALU loop control, and
  // guarantees EXEC is all-1s at every WMMA issue (ISA 7.12 requirement).
  const int wave = __builtin_amdgcn_readfirstlane(tid >> 5);
  const int lane = tid & 31;
  const int b0   = blockIdx.x * BT;

  // ---- load & convert x tile, zero U^T, build triangular scatter table ---
  for (int i = tid; i < BT * BDIM; i += 256)
    xs[i] = (_Float16)x[(size_t)(b0 + (i >> 6)) * BDIM + (i & 63)];
  {
    uint4 z; z.x = z.y = z.z = z.w = 0u;
    uint4* Uq = (uint4*)Ut;
    for (int i = tid; i < (BT * BDIM * BDIM) / 8; i += 256) Uq[i] = z;
  }
  for (int n = tid; n < TDIM; n += 256) {
    int i, j;
    if (n < BDIM) {
      i = n; j = n;
    } else {
      int idx = n - BDIM;                       // 0..2015, s(i)=i*(127-i)/2
      i = (int)((127.0f - sqrtf((float)(16129 - 8 * idx))) * 0.5f);
      if (i < 0) i = 0;
      if (i > 62) i = 62;
      while (i < 62 && (i + 1) * (127 - (i + 1)) / 2 <= idx) ++i;
      while (i > 0 && i * (127 - i) / 2 > idx) --i;
      j = i + 1 + idx - i * (127 - i) / 2;
    }
    ijtab[n] = (unsigned short)((j << 8) | i);
  }
  __syncthreads();

  // ---- stage 1: h1 = tanh(x @ W1 + b1)   [16x64]x[64x256] ----------------
  for (int nt = wave; nt < HDIM / 16; nt += NWAVE) {
    v8f acc0 = {}, acc1 = {};
    acc0 = wmma16(load_a(xs, BDIM, 0, lane),
                  load_b(W1t, BDIM, nt * 16, 0, lane), acc0);
    acc1 = wmma16(load_a(xs, BDIM, 32, lane),
                  load_b(W1t, BDIM, nt * 16, 32, lane), acc1);
    v8f acc;
#pragma unroll
    for (int i = 0; i < 8; ++i) acc[i] = acc0[i] + acc1[i];
    store_h(h1, HDIM, nt * 16, acc, b1, lane);
  }
  __syncthreads();

  // ---- stage 2: h2 = tanh(h1 @ W2 + b2)  [16x256]x[256x256] --------------
  for (int nt = wave; nt < HDIM / 16; nt += NWAVE) {
    v8f acc0 = {}, acc1 = {};   // dual accumulators: independent WMMA chains
#pragma unroll
    for (int kt = 0; kt < HDIM / 32; kt += 2) {
      acc0 = wmma16(load_a(h1, HDIM, kt * 32, lane),
                    load_b(W2t, HDIM, nt * 16, kt * 32, lane), acc0);
      acc1 = wmma16(load_a(h1, HDIM, (kt + 1) * 32, lane),
                    load_b(W2t, HDIM, nt * 16, (kt + 1) * 32, lane), acc1);
    }
    v8f acc;
#pragma unroll
    for (int i = 0; i < 8; ++i) acc[i] = acc0[i] + acc1[i];
    store_h(h2, HDIM, nt * 16, acc, b2, lane);
  }
  __syncthreads();

  // ---- stage 3: t = tanh(h2 @ W3 + b3), scattered into U^T ---------------
  for (int nt = wave; nt < TDIM / 16; nt += NWAVE) {
    // prefetch next B panel for this wave
    if (nt + NWAVE < TDIM / 16)
      __builtin_prefetch(W3t + (size_t)(nt + NWAVE) * 16 * HDIM, 0, 1);
    v8f acc0 = {}, acc1 = {};
#pragma unroll
    for (int kt = 0; kt < HDIM / 32; kt += 2) {
      acc0 = wmma16(load_a(h2, HDIM, kt * 32, lane),
                    load_b(W3t, HDIM, nt * 16, kt * 32, lane), acc0);
      acc1 = wmma16(load_a(h2, HDIM, (kt + 1) * 32, lane),
                    load_b(W3t, HDIM, nt * 16, (kt + 1) * 32, lane), acc1);
    }

    int n  = nt * 16 + (lane & 15);
    float bv = b3[n];
    int packed = ijtab[n];
    int i = packed & 255;
    int j = packed >> 8;
    int mb = (lane >> 4) << 3;
#pragma unroll
    for (int r = 0; r < 8; ++r) {
      int m = mb + r;
      // U[i][j] stored transposed: Ut[m][j][i]
      Ut[((m << 6) + j) * BDIM + i] =
          (_Float16)fast_tanh(acc0[r] + acc1[r] + bv);
    }
  }
  __syncthreads();

  // ---- stage 4: y_b = U_b^T @ U_b  (per-row 64x64x64 GEMM) ---------------
  // A[mm][kk] = Ut[m0+mm][k0+kk] (row-major in Ut), B[kk][nn] = Ut[n0+nn][k0+kk]
  // -> both operands read contiguous K runs from the same Ut buffer.
  for (int job = wave; job < BT * 16; job += NWAVE) {
    int row = job >> 4;
    int tl  = job & 15;
    int mt  = tl >> 2;
    int ntl = tl & 3;
    const _Float16* Um = Ut + row * BDIM * BDIM;
    v8f acc0 = {}, acc1 = {};
    acc0 = wmma16(load_a(Um + mt * 16 * BDIM, BDIM, 0, lane),
                  load_b(Um, BDIM, ntl * 16, 0, lane), acc0);
    acc1 = wmma16(load_a(Um + mt * 16 * BDIM, BDIM, 32, lane),
                  load_b(Um, BDIM, ntl * 16, 32, lane), acc1);

    int n = ntl * 16 + (lane & 15);
    float* yb = y + (size_t)(b0 + row) * (BDIM * BDIM);
    int mb = mt * 16 + ((lane >> 4) << 3);
#pragma unroll
    for (int r = 0; r < 8; ++r)
      yb[(mb + r) * BDIM + n] = acc0[r] + acc1[r];
  }
}

extern "C" void kernel_launch(void* const* d_in, const int* in_sizes, int n_in,
                              void* d_out, int out_size, void* d_ws, size_t ws_size,
                              hipStream_t stream) {
  const float* x  = (const float*)d_in[0];
  const float* W1 = (const float*)d_in[1];
  const float* b1 = (const float*)d_in[2];
  const float* W2 = (const float*)d_in[3];
  const float* b2 = (const float*)d_in[4];
  const float* W3 = (const float*)d_in[5];
  const float* b3 = (const float*)d_in[6];
  float* y = (float*)d_out;

  // Workspace: transposed f16 weights (B-tiles become contiguous b128 loads)
  _Float16* W1t = (_Float16*)d_ws;              // 256*64
  _Float16* W2t = W1t + HDIM * BDIM;            // 256*256
  _Float16* W3t = W2t + HDIM * HDIM;            // 2080*256

  int t1 = BDIM * HDIM, t2 = HDIM * HDIM, t3 = HDIM * TDIM;
  cvt_transpose_f16<<<(t1 + 255) / 256, 256, 0, stream>>>(W1, W1t, BDIM, HDIM);
  cvt_transpose_f16<<<(t2 + 255) / 256, 256, 0, stream>>>(W2, W2t, HDIM, HDIM);
  cvt_transpose_f16<<<(t3 + 255) / 256, 256, 0, stream>>>(W3, W3t, HDIM, TDIM);

  int nblocks = 32768 / BT;  // 2048 workgroups of 16 batch rows
  hm_fused<<<nblocks, 256, 0, stream>>>(x, W1t, b1, W2t, b2, W3t, b3, y);
}
